// DonutSwinSelfAttention_73005854097780
// MI455X (gfx1250) — matrix operand order repaired and checked by reading, not compiled, not run on hardware
//
#include <hip/hip_runtime.h>
#include <hip/hip_bf16.h>

// ---------------------------------------------------------------------------
// DonutSwin window self-attention, MI455X (gfx1250), wave32 + WMMA bf16.
//
// Roofline: ~50 GFLOP vs >=410 MB unavoidable HBM traffic (hidden in, out out)
// -> memory bound (~18 us @ 23.3 TB/s) ONLY if fully fused. One block per
// window: QKV projection + bias + mask + softmax + PV all in LDS, weights
// pre-packed ONCE to bf16 in d_ws (L2-resident, half the per-block bytes, no
// per-block f32->bf16 conversion VALU). All matmuls: v_wmma_f32_16x16x32_bf16.
// ---------------------------------------------------------------------------

typedef __attribute__((ext_vector_type(16))) __bf16 v16bf;
typedef __attribute__((ext_vector_type(8)))  __bf16 v8bf;
typedef __attribute__((ext_vector_type(8)))  float  v8f;

#define NW   49      // tokens per window
#define CDIM 256     // hidden dim
#define NH   8       // heads
#define DHD  32      // head dim
#define BWIN 4096    // windows
#define MWIN 64      // mask leading dim
#define NOUT 384     // fused QKV output features (256 Q | 64 K | 64 V)

// LDS layout (bytes). Strides padded for bank spread + 16B alignment of all
// b128 LDS loads used as WMMA operands.
#define HB_STRIDE 264                     // hbuf row stride (halves)
#define QS_STRIDE 264                     // qsm row stride (halves)
#define KS_STRIDE 40                      // ksm row stride (halves)
#define VS_STRIDE 72                      // vsm row stride (halves)
#define SC_STRIDE 68                      // score row stride (floats), 272B

#define HBUF_OFF 0
#define QSM_OFF  25888                    // 49*264*2 = 25872, padded
#define KSM_OFF  59680                    // + 64*264*2 = 33792
#define VSM_OFF  69920                    // + 2*64*40*2 = 10240
#define SC_OFF   79136                    // + 2*32*72*2 = 9216
#define SMEM_TOTAL 218400                 // + 8*64*68*4 = 139264  (<320KB/WG)

// Workspace layout (bytes)
#define WS_BIASH_OFF 0                    // 8*49*49 f32 = 76832
#define WS_WPACK_OFF 76928                // 384*256 bf16 = 196608
#define WS_BVEC_OFF  273536               // 384 f32 = 1536

static __device__ __forceinline__ v16bf cat8(v8bf lo, v8bf hi) {
  return __builtin_shufflevector(lo, hi, 0,1,2,3,4,5,6,7,8,9,10,11,12,13,14,15);
}
static __device__ __forceinline__ v8f vzero8() {
  v8f z;
  #pragma unroll
  for (int i = 0; i < 8; ++i) z[i] = 0.0f;
  return z;
}
static __device__ __forceinline__ v8f wmma_bf16(v16bf a, v16bf b, v8f c) {
  // (neg_a, A, neg_b, B, c_mod, C, reuse_a, reuse_b)
  return __builtin_amdgcn_wmma_f32_16x16x32_bf16(false, a, false, b, (short)0, c,
                                                 false, false);
}

// Pre-gather relative-position bias per head: biasH[h][q][k] (8*49*49 f32;
// tiny, L2-resident, makes the softmax inner loop contiguous).
__global__ void swin_bias_gather(const float* __restrict__ table,
                                 const int* __restrict__ idx,
                                 float* __restrict__ biasH) {
  int t = blockIdx.x * 256 + threadIdx.x;
  if (t >= NH * NW * NW) return;
  int h  = t / (NW * NW);
  int qk = t - h * (NW * NW);
  biasH[t] = table[idx[qk] * NH + h];
}

// Pack [Wq; Wk; Wv] -> bf16 row-major [384][256] once (dispatch-invariant),
// plus fused bias vector [384] f32. Served from L2 to all 4096 blocks.
__global__ void swin_pack_weights(const float* __restrict__ qw,
                                  const float* __restrict__ kw,
                                  const float* __restrict__ vw,
                                  const float* __restrict__ qb,
                                  const float* __restrict__ kb,
                                  const float* __restrict__ vb,
                                  __bf16* __restrict__ wpack,
                                  float* __restrict__ bpack) {
  int t = blockIdx.x * 256 + threadIdx.x;       // 0 .. 384*256-1
  if (t >= NOUT * CDIM) return;
  const int o = t >> 8, c = t & 255;
  float v;
  if (o < CDIM)      v = qw[o * CDIM + c];
  else if (o < 320)  v = kw[(o - 256) * CDIM + c];
  else               v = vw[(o - 320) * CDIM + c];
  wpack[t] = (__bf16)v;
  if (c == 0)
    bpack[o] = (o < CDIM) ? qb[o] : (o < 320 ? kb[o - 256] : vb[o - 320]);
}

__global__ void __launch_bounds__(256, 1)
swin_attn_fused(const float* __restrict__ hidden,
                const float* __restrict__ mask,
                const __bf16* __restrict__ wpack,
                const float* __restrict__ bpack,
                const float* __restrict__ biasH,
                float* __restrict__ out) {
  extern __shared__ char smem[];
  __bf16* hbuf = (__bf16*)(smem + HBUF_OFF);  // [49][264]  hidden window, bf16
  __bf16* qsm  = (__bf16*)(smem + QSM_OFF);   // [64][264]  Q  (token-major)
  __bf16* ksm  = (__bf16*)(smem + KSM_OFF);   // [2][64][40] K (token-major)
  __bf16* vsm  = (__bf16*)(smem + VSM_OFF);   // [2][32][72] V^T (dh-major)

  const int b    = blockIdx.x;
  const int tid  = threadIdx.x;
  const int wave = tid >> 5;                  // wave32: 8 waves/block
  const int lane = tid & 31;
  const int lh   = lane & 15;
  const int hi16 = (lane >= 16);
  const int aoff = hi16 ? 8 : 0;              // A-operand K pattern {0-7,16-23}/{8-15,24-31}
  const int boff = hi16 ? 8 : 0;              // B-operand K pattern {0-7,16-23}/{8-15,24-31}

  const float* hwin = hidden + (size_t)b * (NW * CDIM);

  // ---- Phase 0: zero V^T pad (avoid NaN*0 in PV), stage hidden -> bf16 LDS
  for (int i = tid; i < 2 * DHD * VS_STRIDE; i += 256) vsm[i] = (__bf16)0.0f;
  for (int e4 = tid * 4; e4 < NW * CDIM; e4 += 256 * 4) {
    float4 v = *(const float4*)(hwin + e4);
    int q = e4 >> 8, c = e4 & 255;
    __bf16* d = hbuf + q * HB_STRIDE + c;
    d[0] = (__bf16)v.x; d[1] = (__bf16)v.y; d[2] = (__bf16)v.z; d[3] = (__bf16)v.w;
  }
  __syncthreads();

  // ---- Phase 1: fused QKV projection. M=64(pad of 49), N=384 (Q|K|V), K=256.
  // Each wave owns 3 n-tiles x 4 m-tiles; bf16 weights stream from L2.
  #pragma unroll
  for (int j = 0; j < 3; ++j) {
    const int nt = wave * 3 + j;
    const int o  = nt * 16 + lh;              // output feature column
    const __bf16* wrow = wpack + (size_t)o * CDIM;
    const float bias = bpack[o];

    v16bf Bw[8];
    #pragma unroll
    for (int kc = 0; kc < 8; ++kc) {
      const __bf16* p = wrow + kc * 32 + boff;
      Bw[kc] = cat8(*(const v8bf*)p, *(const v8bf*)(p + 16));
    }

    #pragma unroll
    for (int mt = 0; mt < 4; ++mt) {
      int row = mt * 16 + lh; if (row > NW - 1) row = NW - 1;   // clamp pad rows
      v8f acc = vzero8();
      #pragma unroll
      for (int kc = 0; kc < 8; ++kc) {
        const __bf16* hp = hbuf + row * HB_STRIDE + kc * 32 + aoff;
        acc = wmma_bf16(cat8(*(const v8bf*)hp, *(const v8bf*)(hp + 16)),
                        Bw[kc], acc);
      }
      const int mbase = mt * 16 + (hi16 ? 8 : 0);
      #pragma unroll
      for (int r = 0; r < 8; ++r) {
        const int tok = mbase + r;
        if (tok >= NW) continue;
        const __bf16 bv = (__bf16)(acc[r] + bias);
        if (o < CDIM) {
          qsm[tok * QS_STRIDE + o] = bv;
        } else if (o < 320) {
          const int kvh = (o - 256) >> 5, dh = (o - 256) & 31;
          ksm[(kvh * 64 + tok) * KS_STRIDE + dh] = bv;
        } else {                                    // V stored transposed
          const int kvh = (o - 320) >> 5, dh = (o - 320) & 31;
          vsm[(kvh * DHD + dh) * VS_STRIDE + tok] = bv;
        }
      }
    }
  }
  __syncthreads();

  // ---- Phase 2: per-wave head attention. GQA: kv head = h & 1 (tile order).
  const int h   = wave;
  const int kvh = h & 1;
  float* sc = (float*)(smem + SC_OFF) + h * (64 * SC_STRIDE);  // [64][68] f32

  v16bf qa[4];
  #pragma unroll
  for (int mt = 0; mt < 4; ++mt) {
    int row = mt * 16 + lh; if (row > NW - 1) row = NW - 1;
    const __bf16* qp = qsm + row * QS_STRIDE + h * DHD + aoff;
    qa[mt] = cat8(*(const v8bf*)qp, *(const v8bf*)(qp + 16));
  }
  v16bf kb4[4];
  #pragma unroll
  for (int nt = 0; nt < 4; ++nt) {
    const int tok = nt * 16 + lh;                       // pad cols never read
    const __bf16* kp = ksm + (kvh * 64 + tok) * KS_STRIDE + boff;
    kb4[nt] = cat8(*(const v8bf*)kp, *(const v8bf*)(kp + 16));
  }

  const float scale = 0.17677669529663687f;             // 32^-0.5
  #pragma unroll
  for (int mt = 0; mt < 4; ++mt) {
    #pragma unroll
    for (int nt = 0; nt < 4; ++nt) {
      v8f acc = wmma_bf16(qa[mt], kb4[nt], vzero8());
      const int qrow = mt * 16 + (hi16 ? 8 : 0);
      const int col  = nt * 16 + lh;
      #pragma unroll
      for (int r = 0; r < 8; ++r)
        sc[(qrow + r) * SC_STRIDE + col] = acc[r] * scale;
    }
  }
  // Intra-wave LDS RAW: DS ops are in-order per wave; score buffer is private
  // to this wave's head, so no block barrier needed here.

  // ---- Softmax (f32), + mask + rel-pos bias; write P as bf16 aliased in-row.
  const float* mrow0 = mask  + (size_t)(b & (MWIN - 1)) * (NW * NW);
  const float* brow0 = biasH + (size_t)h * (NW * NW);
  for (int qi = lane; qi < NW; qi += 32) {              // rows 0..48
    float* row = sc + qi * SC_STRIDE;
    const float* mr = mrow0 + qi * NW;
    const float* br = brow0 + qi * NW;
    float mx = -1e30f;
    for (int c = 0; c < NW; ++c) {
      float s = row[c] + mr[c] + br[c];
      row[c] = s; mx = fmaxf(mx, s);
    }
    float sum = 0.0f;
    for (int c = 0; c < NW; ++c) { float e = __expf(row[c] - mx); row[c] = e; sum += e; }
    const float inv = 1.0f / sum;
    __bf16* pr = (__bf16*)row;                          // alias: 2c <= 4c, safe ascending
    for (int c = 0; c < NW; ++c) pr[c] = (__bf16)(row[c] * inv);
    for (int c = NW; c < 64; ++c) pr[c] = (__bf16)0.0f; // zero pad keys -> exact PV
  }

  // ---- P(64x64 bf16) x V^T  ->  out head slice, f32 accum.
  v8f oacc[4][2];
  #pragma unroll
  for (int mt = 0; mt < 4; ++mt) { oacc[mt][0] = vzero8(); oacc[mt][1] = vzero8(); }
  #pragma unroll
  for (int kc = 0; kc < 2; ++kc) {
    v16bf vbop[2];
    #pragma unroll
    for (int nt = 0; nt < 2; ++nt) {
      const int dh = nt * 16 + lh;
      const __bf16* vp = vsm + (kvh * DHD + dh) * VS_STRIDE + kc * 32 + boff;
      vbop[nt] = cat8(*(const v8bf*)vp, *(const v8bf*)(vp + 16));
    }
    #pragma unroll
    for (int mt = 0; mt < 4; ++mt) {
      int row = mt * 16 + lh; if (row > NW - 1) row = NW - 1;
      const __bf16* pp = (const __bf16*)(sc + row * SC_STRIDE) + kc * 32 + aoff;
      v16bf pa = cat8(*(const v8bf*)pp, *(const v8bf*)(pp + 16));
      oacc[mt][0] = wmma_bf16(pa, vbop[0], oacc[mt][0]);
      oacc[mt][1] = wmma_bf16(pa, vbop[1], oacc[mt][1]);
    }
  }

  // ---- Stage head output over the (now fully consumed) score buffer, then
  //      block-wide coalesced float4 store of out[b] (205 MB total writes).
  float* stg = sc;                                      // [64][32] f32 staging
  #pragma unroll
  for (int mt = 0; mt < 4; ++mt) {
    const int qrow = mt * 16 + (hi16 ? 8 : 0);
    #pragma unroll
    for (int nt = 0; nt < 2; ++nt) {
      const int col = nt * 16 + lh;
      #pragma unroll
      for (int r = 0; r < 8; ++r) stg[(qrow + r) * 32 + col] = oacc[mt][nt][r];
    }
  }
  __syncthreads();

  float* owin = out + (size_t)b * (NW * CDIM);
  const float* scb = (const float*)(smem + SC_OFF);
  for (int e4 = tid * 4; e4 < NW * CDIM; e4 += 256 * 4) {
    const int q = e4 >> 8, c = e4 & 255;
    const int hh = c >> 5, col = c & 31;
    float4 v = *(const float4*)(scb + hh * (64 * SC_STRIDE) + q * 32 + col);
    *(float4*)(owin + e4) = v;
  }
}

extern "C" void kernel_launch(void* const* d_in, const int* in_sizes, int n_in,
                              void* d_out, int out_size, void* d_ws, size_t ws_size,
                              hipStream_t stream) {
  const float* hidden = (const float*)d_in[0];
  const float* mask   = (const float*)d_in[1];
  const float* qw     = (const float*)d_in[2];
  const float* qb     = (const float*)d_in[3];
  const float* kw     = (const float*)d_in[4];
  const float* kb     = (const float*)d_in[5];
  const float* vw     = (const float*)d_in[6];
  const float* vb     = (const float*)d_in[7];
  const float* table  = (const float*)d_in[8];
  const int*   idx    = (const int*)d_in[9];
  float* out = (float*)d_out;

  char* ws = (char*)d_ws;
  float*  biasH = (float*)(ws + WS_BIASH_OFF);   // [8][49][49] f32
  __bf16* wpack = (__bf16*)(ws + WS_WPACK_OFF);  // [384][256] bf16
  float*  bpack = (float*)(ws + WS_BVEC_OFF);    // [384] f32

  swin_bias_gather<<<(NH * NW * NW + 255) / 256, 256, 0, stream>>>(table, idx, biasH);
  swin_pack_weights<<<(NOUT * CDIM + 255) / 256, 256, 0, stream>>>(qw, kw, vw,
                                                                   qb, kb, vb,
                                                                   wpack, bpack);
  swin_attn_fused<<<BWIN, 256, SMEM_TOTAL, stream>>>(hidden, mask, wpack, bpack,
                                                     biasH, out);
}
